// RNN_wavefunction_18648747999729
// MI455X (gfx1250) — matrix operand order repaired and checked by reading, not compiled, not run on hardware
//
#include <hip/hip_runtime.h>
#include <hip/hip_bf16.h>
#include <math.h>

typedef _Float16 h16;
typedef __attribute__((ext_vector_type(16))) _Float16 v16h;
typedef __attribute__((ext_vector_type(8)))  _Float16 v8h;
typedef __attribute__((ext_vector_type(8)))  float    v8f;

#define NSTEPS 128
#define BSZ    8192
#define HD     128
#define ROWS_PER_WG 64
#define THREADS 128   // 4 waves of 32

#define LOG2E  1.4426950408889634f
#define LN2    0.6931471805599453f

// tanh(x) = 1 - 2/(exp(2x)+1), hardware exp2 + rcp.
// Saturates correctly: x>>0 -> exp2->inf -> 1 ; x<<0 -> exp2->0 -> -1.
static __device__ __forceinline__ float fast_tanh(float x) {
    float e = __builtin_amdgcn_exp2f(x * (2.0f * LOG2E));
    return 1.0f - 2.0f * __builtin_amdgcn_rcpf(e + 1.0f);
}

__global__ __launch_bounds__(THREADS) void rnn_wf_kernel(
    const float* __restrict__ samples,                      // (N,B,2) one-hot
    const float* __restrict__ W_ih0, const float* __restrict__ b_ih0,
    const float* __restrict__ W_hh0, const float* __restrict__ b_hh0,
    const float* __restrict__ W_ih1, const float* __restrict__ b_ih1,
    const float* __restrict__ W_hh1, const float* __restrict__ b_hh1,
    const float* __restrict__ W_dense, const float* __restrict__ b_dense,
    float* __restrict__ out)
{
    // B-fragment-packed weights: [which][tile = nt*4+kc][lane*16 + i]
    // lane's 16 halves = 16 consecutive K for column n = nt*16 + lane%16,
    // K block = kc*32 + (lane>=16 ? 16 : 0).
    __shared__ __align__(16) h16 sW[3][32][512];            // 96 KB
    __shared__ __align__(16) h16 sH0[ROWS_PER_WG * HD];     // 16 KB, row-major f16
    __shared__ __align__(16) h16 sH1[ROWS_PER_WG * HD];     // 16 KB
    __shared__ float sBias0[HD], sBias1[HD], sWa[HD], sWdlt[HD], sD0[HD], sD1[HD];

    const int tid = threadIdx.x;

    // ---- one-time packing of weights into WMMA-B layout (f32 -> f16) ----
    {
        const float* wsrc[3] = { W_hh0, W_ih1, W_hh1 };     // B[k][n] = W[n][k]
        #pragma unroll
        for (int w = 0; w < 3; ++w) {
            for (int idx = tid; idx < 32 * 32 * 16; idx += THREADS) {
                const int i    = idx & 15;
                const int ln   = (idx >> 4) & 31;
                const int tile = idx >> 9;
                const int nt = tile >> 2, kc = tile & 3;
                const int k = kc * 32 + (ln >> 4) * 16 + i;
                const int n = nt * 16 + (ln & 15);
                sW[w][tile][ln * 16 + i] = (h16)wsrc[w][n * HD + k];
            }
        }
    }
    for (int n = tid; n < HD; n += THREADS) {
        sBias0[n] = b_ih0[n] + b_hh0[n];
        sBias1[n] = b_ih1[n] + b_hh1[n];
        sWa[n]   = W_ih0[n * 2 + 0];                        // column for bit==0
        sWdlt[n] = W_ih0[n * 2 + 1] - W_ih0[n * 2 + 0];     // delta for bit==1
        sD0[n] = W_dense[n];
        sD1[n] = W_dense[HD + n];
    }
    for (int idx = tid; idx < ROWS_PER_WG * HD; idx += THREADS) {
        sH0[idx] = (h16)0.f; sH1[idx] = (h16)0.f;           // h = 0 at t=0
    }
    __syncthreads();

    const int lane  = tid & 31;
    const int hl    = lane & 15;        // column-within-tile / A row
    const int hh    = lane >> 4;        // half select
    const int wrow0 = (tid >> 5) * 16;  // wave's first row within WG
    const int gbase = blockIdx.x * ROWS_PER_WG + wrow0;

    const float bd0 = b_dense[0], bd1 = b_dense[1];
    float lp = 0.f;

    for (int t = 0; t < NSTEPS; ++t) {
        // previous-step spin for row (gbase+hl); broadcast to rows via shfl
        const int tm1 = (t > 0) ? (t - 1) : 0;
        const float bv = samples[tm1 * (BSZ * 2) + (gbase + hl) * 2 + 1];
        const float tprev = (t > 0) ? 1.f : 0.f;
        float bitf[8];
        #pragma unroll
        for (int r = 0; r < 8; ++r) bitf[r] = __shfl(bv, r + 8 * hh, 32);

        // ================= layer 0: acc = h0 @ W_hh0^T =================
        v16h a0[4];
        #pragma unroll
        for (int c = 0; c < 4; ++c) {
            const h16* p = &sH0[(wrow0 + hl) * HD + c * 32 + hh * 16];
            v8h lo = *(const v8h*)p;
            v8h hi = *(const v8h*)(p + 8);
            #pragma unroll
            for (int i = 0; i < 8; ++i) { a0[c][i] = lo[i]; a0[c][i + 8] = hi[i]; }
        }
        v8f acc[8];
        #pragma unroll
        for (int nt = 0; nt < 8; ++nt) {
            v8f c = {};
            #pragma unroll
            for (int kc = 0; kc < 4; ++kc) {
                const h16* p = &sW[0][nt * 4 + kc][lane * 16];
                v8h lo = *(const v8h*)p;
                v8h hi = *(const v8h*)(p + 8);
                v16h b;
                #pragma unroll
                for (int i = 0; i < 8; ++i) { b[i] = lo[i]; b[i + 8] = hi[i]; }
                c = __builtin_amdgcn_wmma_f32_16x16x32_f16(
                        false, a0[kc], false, b, (short)0, c, false, false);
            }
            acc[nt] = c;
        }
        // epilogue: + bias + one-hot input term, tanh, store new h0 (f16)
        #pragma unroll
        for (int nt = 0; nt < 8; ++nt) {
            const int nc = nt * 16 + hl;
            const float wbase = sBias0[nc] + tprev * sWa[nc];   // bias + W_ih0[:,0]
            const float wdel  = tprev * sWdlt[nc];
            #pragma unroll
            for (int r = 0; r < 8; ++r) {
                float v = fast_tanh(acc[nt][r] + wbase + bitf[r] * wdel);
                sH0[(wrow0 + r + 8 * hh) * HD + nc] = (h16)v;
            }
        }
        asm volatile("s_wait_dscnt 0" ::: "memory");

        // ======= layer 1: acc2 = h0' @ W_ih1^T + h1 @ W_hh1^T =======
        v16h a1[4], a0n[4];
        #pragma unroll
        for (int c = 0; c < 4; ++c) {
            const h16* p1 = &sH1[(wrow0 + hl) * HD + c * 32 + hh * 16];
            const h16* p0 = &sH0[(wrow0 + hl) * HD + c * 32 + hh * 16];
            v8h lo1 = *(const v8h*)p1, hi1 = *(const v8h*)(p1 + 8);
            v8h lo0 = *(const v8h*)p0, hi0 = *(const v8h*)(p0 + 8);
            #pragma unroll
            for (int i = 0; i < 8; ++i) {
                a1[c][i] = lo1[i];  a1[c][i + 8] = hi1[i];
                a0n[c][i] = lo0[i]; a0n[c][i + 8] = hi0[i];
            }
        }
        v8f acc2[8];
        #pragma unroll
        for (int nt = 0; nt < 8; ++nt) {
            v8f c = {};
            #pragma unroll
            for (int kc = 0; kc < 4; ++kc) {
                const h16* p = &sW[1][nt * 4 + kc][lane * 16];
                v8h lo = *(const v8h*)p; v8h hi = *(const v8h*)(p + 8);
                v16h b;
                #pragma unroll
                for (int i = 0; i < 8; ++i) { b[i] = lo[i]; b[i + 8] = hi[i]; }
                c = __builtin_amdgcn_wmma_f32_16x16x32_f16(
                        false, a0n[kc], false, b, (short)0, c, false, false);
            }
            #pragma unroll
            for (int kc = 0; kc < 4; ++kc) {
                const h16* p = &sW[2][nt * 4 + kc][lane * 16];
                v8h lo = *(const v8h*)p; v8h hi = *(const v8h*)(p + 8);
                v16h b;
                #pragma unroll
                for (int i = 0; i < 8; ++i) { b[i] = lo[i]; b[i + 8] = hi[i]; }
                c = __builtin_amdgcn_wmma_f32_16x16x32_f16(
                        false, a1[kc], false, b, (short)0, c, false, false);
            }
            acc2[nt] = c;
        }
        // epilogue: tanh, store h1', and partial dense dot products
        float pA[8], pB[8];
        #pragma unroll
        for (int r = 0; r < 8; ++r) { pA[r] = 0.f; pB[r] = 0.f; }
        #pragma unroll
        for (int nt = 0; nt < 8; ++nt) {
            const int nc = nt * 16 + hl;
            const float bs = sBias1[nc], w0 = sD0[nc], w1 = sD1[nc];
            #pragma unroll
            for (int r = 0; r < 8; ++r) {
                float v = fast_tanh(acc2[nt][r] + bs);
                sH1[(wrow0 + r + 8 * hh) * HD + nc] = (h16)v;
                pA[r] += v * w0;
                pB[r] += v * w1;
            }
        }
        // dense + log-softmax per row (reduce over 16-lane halves)
        #pragma unroll
        for (int r = 0; r < 8; ++r) {
            float s0 = pA[r], s1 = pB[r];
            #pragma unroll
            for (int mk = 1; mk < 16; mk <<= 1) {
                s0 += __shfl_xor(s0, mk, 32);
                s1 += __shfl_xor(s1, mk, 32);
            }
            if (hl == r) {
                const float l0 = s0 + bd0, l1 = s1 + bd1;
                const float mx = fmaxf(l0, l1);
                const float ed = __builtin_amdgcn_exp2f(-fabsf(l0 - l1) * LOG2E);
                const float lse = mx + LN2 * __builtin_amdgcn_logf(1.0f + ed);
                const int rowg = gbase + r + 8 * hh;
                const float sc = samples[t * (BSZ * 2) + rowg * 2 + 1];
                lp += (l0 + sc * (l1 - l0)) - lse;
            }
        }
    }

    if (hl < 8) out[gbase + hl + 8 * hh] = lp;   // (1,B) flat
}

extern "C" void kernel_launch(void* const* d_in, const int* in_sizes, int n_in,
                              void* d_out, int out_size, void* d_ws, size_t ws_size,
                              hipStream_t stream) {
    const float* samples = (const float*)d_in[0];
    const float* W_ih0   = (const float*)d_in[1];
    const float* b_ih0   = (const float*)d_in[2];
    const float* W_hh0   = (const float*)d_in[3];
    const float* b_hh0   = (const float*)d_in[4];
    const float* W_ih1   = (const float*)d_in[5];
    const float* b_ih1   = (const float*)d_in[6];
    const float* W_hh1   = (const float*)d_in[7];
    const float* b_hh1   = (const float*)d_in[8];
    const float* W_dense = (const float*)d_in[9];
    const float* b_dense = (const float*)d_in[10];
    float* out = (float*)d_out;

    dim3 grid(BSZ / ROWS_PER_WG);   // 128 workgroups
    dim3 block(THREADS);            // 4 waves
    rnn_wf_kernel<<<grid, block, 0, stream>>>(
        samples, W_ih0, b_ih0, W_hh0, b_hh0,
        W_ih1, b_ih1, W_hh1, b_hh1, W_dense, b_dense, out);
}
